// class_cross_MultiHeadAttention_54065048322416
// MI455X (gfx1250) — matrix operand order repaired and checked by reading, not compiled, not run on hardware
//
#include <hip/hip_runtime.h>

#define D_MODEL 256
#define NHEAD 4
#define DPH 64
#define NLBL 9

typedef __attribute__((ext_vector_type(16))) __bf16 v16bf;
typedef __attribute__((ext_vector_type(8)))  __bf16 v8bf;
typedef __attribute__((ext_vector_type(8)))  float  v8f;

typedef __attribute__((ext_vector_type(4))) unsigned int u32x4;
typedef __attribute__((ext_vector_type(8))) int          i32x8;
typedef __attribute__((ext_vector_type(4))) int          i32x4;

union BF16x16 { v16bf v; v8bf h[2]; __bf16 e[16]; };
union F32x8   { v8f  v; float  e[8];  };

#if defined(__AMDGCN__) && defined(__has_builtin)
#if __has_builtin(__builtin_amdgcn_tensor_load_to_lds)
#define HAVE_TDM 1
#endif
#endif
#ifndef HAVE_TDM
#define HAVE_TDM 0
#endif

// ---------------------------------------------------------------------------
// TDM: DMA a 2D bf16 tile (tile_h rows x tile_w elems, row stride in elems)
// from global into LDS byte offset lds_off (D# per CDNA5 ISA).
// ---------------------------------------------------------------------------
__device__ __forceinline__
void tdm_load_tile_bf16(const __bf16* gptr, unsigned lds_off,
                        unsigned tile_w, unsigned tile_h, unsigned stride_elems)
{
#if HAVE_TDM
    unsigned long long ga = (unsigned long long)gptr;
    u32x4 g0;
    g0[0] = 1u;                                          // count=1, user desc
    g0[1] = lds_off;                                     // lds_addr (bytes)
    g0[2] = (unsigned)ga;                                // global_addr[31:0]
    g0[3] = (unsigned)((ga >> 32) & 0x01ffffffu)         // global_addr[56:32]
          | (2u << 30);                                  // type = 2 ("image")
    const unsigned dim0 = 0x40000000u;                   // huge: no OOB clip
    const unsigned dim1 = 0x40000000u;
    i32x8 g1;
    g1[0] = (int)(1u << 16);                             // data_size = 2 bytes
    g1[1] = (int)((dim0 & 0xffffu) << 16);               // tensor_dim0 lo
    g1[2] = (int)((dim0 >> 16) | ((dim1 & 0xffffu) << 16));
    g1[3] = (int)((dim1 >> 16) | (tile_w << 16));        // tile_dim0
    g1[4] = (int)(tile_h & 0xffffu);                     // tile_dim1 (dim2=0)
    g1[5] = (int)stride_elems;                           // tensor_dim0_stride
    g1[6] = 0;
    g1[7] = 0;
    i32x4 z4 = {0, 0, 0, 0};
#if __clang_major__ >= 23
    i32x8 z8 = {0, 0, 0, 0, 0, 0, 0, 0};
    __builtin_amdgcn_tensor_load_to_lds(g0, g1, z4, z4, z8, 0);
#else
    __builtin_amdgcn_tensor_load_to_lds(g0, g1, z4, z4, 0);
#endif
#endif
}

// ---------------------------------------------------------------------------
// Kernel 0a: elementwise f32 -> bf16 (8 elems/thread, 128-bit stores).
// ---------------------------------------------------------------------------
__global__ __launch_bounds__(256)
void cvt_bf16_kernel(const float* __restrict__ src, __bf16* __restrict__ dst,
                     int n)
{
    int idx = (blockIdx.x * 256 + threadIdx.x) * 8;
    if (idx >= n) return;
    float4 a = *(const float4*)(src + idx);
    float4 b = *(const float4*)(src + idx + 4);
    v8bf o;
    o[0] = (__bf16)a.x; o[1] = (__bf16)a.y; o[2] = (__bf16)a.z; o[3] = (__bf16)a.w;
    o[4] = (__bf16)b.x; o[5] = (__bf16)b.y; o[6] = (__bf16)b.z; o[7] = (__bf16)b.w;
    *(v8bf*)(dst + idx) = o;
}

// ---------------------------------------------------------------------------
// Kernel 0b: per-slice 256x256 transpose + f32->bf16:
//   dst[z][o][k] = (bf16) src[z][k][o]
// grid = (16, 16, nslices), 32 threads; 128-bit bf16 stores.
// ---------------------------------------------------------------------------
__global__ __launch_bounds__(32)
void transpose_cvt_kernel(const float* __restrict__ src,
                          __bf16* __restrict__ dst)
{
    const int lane = threadIdx.x;
    const int o  = lane >> 1;              // 0..15  (output row within tile)
    const int kh = (lane & 1) * 8;         // 0 or 8 (k half within tile)
    const size_t base = (size_t)blockIdx.z * D_MODEL * D_MODEL;
    const int r0 = blockIdx.x * 16;        // k tile
    const int c0 = blockIdx.y * 16;        // o tile
    v8bf outv;
#pragma unroll
    for (int j = 0; j < 8; ++j)
        outv[j] = (__bf16)src[base + (size_t)(r0 + kh + j) * D_MODEL + c0 + o];
    *(v8bf*)(dst + base + (size_t)(c0 + o) * D_MODEL + r0 + kh) = outv;
}

// ---------------------------------------------------------------------------
// Kernel 1: per-label projection, all-bf16 128-bit loads.
// grid = (T/16, D/16, NLBL), 32 threads.  transposeT != 0 => store
// out[col][row] (leading dim T) so attention's P*V B-fragments are contiguous.
// ---------------------------------------------------------------------------
__global__ __launch_bounds__(32)
void proj_label_kernel(const __bf16* __restrict__ x,    // [T, 256] bf16
                       const int*    __restrict__ lbl,  // [T]
                       const __bf16* __restrict__ WT,   // [NLBL, 256, 256] o-major
                       const float*  __restrict__ b,    // [NLBL, 256]
                       __bf16*       __restrict__ out,
                       int T, int transposeT)
{
    const int lane = threadIdx.x;
    const int g  = lane >> 4;
    const int lm = lane & 15;
    const int row0  = blockIdx.x * 16;
    const int col0  = blockIdx.y * 16;
    const int label = blockIdx.z;

    const __bf16* WTl = WT + (size_t)label * D_MODEL * D_MODEL;

    F32x8 acc;
#pragma unroll
    for (int r = 0; r < 8; ++r) acc.e[r] = 0.f;

    for (int k0 = 0; k0 < D_MODEL; k0 += 32) {
        BF16x16 a, bm;
        // A fragment (contiguous k = g*16 + s, same order as B): two b128 loads
        const __bf16* xr = x + (size_t)(row0 + lm) * D_MODEL + k0;
        a.h[0] = *(const v8bf*)(xr + g * 16);
        a.h[1] = *(const v8bf*)(xr + g * 16 + 8);
        // B fragment: B(K,N) = W[k0+K][col0+N] = WT[col0+N][k0+K]; contiguous
        const __bf16* wr = WTl + (size_t)(col0 + lm) * D_MODEL + k0 + g * 16;
        bm.h[0] = *(const v8bf*)(wr);
        bm.h[1] = *(const v8bf*)(wr + 8);

        acc.v = __builtin_amdgcn_wmma_f32_16x16x32_bf16(
            false, a.v, false, bm.v, (short)0, acc.v, false, false);
    }

#pragma unroll
    for (int r = 0; r < 8; ++r) {
        int m = row0 + r + g * 8;
        if (lbl[m] == label) {
            float val = acc.e[r] + b[(size_t)label * D_MODEL + col0 + lm];
            if (!transposeT)
                out[(size_t)m * D_MODEL + col0 + lm] = (__bf16)val;
            else
                out[(size_t)(col0 + lm) * T + m] = (__bf16)val;
        }
    }
}

// ---------------------------------------------------------------------------
// Kernel 2: flash attention with label masking.
// grid = (M/16, NHEAD), 32 threads.  K tiles stream through ping-pong LDS
// buffers via the Tensor Data Mover; V comes from the transposed copy with
// contiguous 128-bit loads.
// ---------------------------------------------------------------------------
__global__ __launch_bounds__(32)
void attn_kernel(const __bf16* __restrict__ q,    // [M,256] bf16
                 const __bf16* __restrict__ k,    // [N,256] bf16
                 const __bf16* __restrict__ vT,   // [256,N] bf16 (transposed)
                 const int*    __restrict__ lbl2, // [M]
                 const int*    __restrict__ lbl1, // [N]
                 __bf16*       __restrict__ ctx,  // [M,256] bf16
                 int M, int N)
{
    __shared__ __attribute__((aligned(16))) __bf16 lds_k[2][32 * 64]; // 2x4 KB
    __shared__ __attribute__((aligned(16))) __bf16 lds_p[16 * 32];    // P restripe

    const int lane = threadIdx.x;
    const int g  = lane >> 4;
    const int lm = lane & 15;
    const int m0 = blockIdx.x * 16;
    const int h  = blockIdx.y;
    const int hd = h * DPH;

    const unsigned ldsk_base =
        (unsigned)(unsigned long long)(const void*)&lds_k[0][0];

    // Q fragments (two 16x32 A fragments), 128-bit loads
    BF16x16 qa[2];
#pragma unroll
    for (int c = 0; c < 2; ++c) {
        const __bf16* qr = q + (size_t)(m0 + lm) * D_MODEL + hd + c * 32;
        qa[c].h[0] = *(const v8bf*)(qr + g * 16);
        qa[c].h[1] = *(const v8bf*)(qr + g * 16 + 8);
    }

    int mylbl[8];
#pragma unroll
    for (int r = 0; r < 8; ++r) mylbl[r] = lbl2[m0 + r + g * 8];

    float rowmax[8], rowsum[8];
#pragma unroll
    for (int r = 0; r < 8; ++r) { rowmax[r] = -1e30f; rowsum[r] = 0.f; }

    F32x8 o[4];
#pragma unroll
    for (int t = 0; t < 4; ++t)
#pragma unroll
        for (int r = 0; r < 8; ++r) o[t].e[r] = 0.f;

#if HAVE_TDM
    // prologue: DMA first K tile (32 rows x 64 head dims) into buffer 0
    tdm_load_tile_bf16(k + hd, ldsk_base, 64, 32, D_MODEL);
#endif

    for (int n0 = 0; n0 < N; n0 += 32) {
        const int buf = (n0 >> 5) & 1;
        const __bf16* kt;
#if HAVE_TDM
        if (n0 + 32 < N) {
            // issue next tile into the other buffer, then wait for this one
            asm volatile("s_wait_dscnt 0x0" ::: "memory");
            tdm_load_tile_bf16(k + (size_t)(n0 + 32) * D_MODEL + hd,
                               ldsk_base + (unsigned)((buf ^ 1) * 4096),
                               64, 32, D_MODEL);
            __builtin_amdgcn_s_wait_tensorcnt(1);   // in-order: tile n0 done
        } else {
            __builtin_amdgcn_s_wait_tensorcnt(0);
        }
        kt = &lds_k[buf][0];
#else
        {   // cooperative fallback: each lane copies one key row
            const __bf16* src = k + (size_t)(n0 + lane) * D_MODEL + hd;
#pragma unroll
            for (int j = 0; j < 8; ++j)
                *(v8bf*)(&lds_k[0][0] + lane * 64 + j * 8) =
                    *(const v8bf*)(src + j * 8);
            __syncthreads();
        }
        kt = &lds_k[0][0];
#endif
        if (n0 + 32 < N)                      // prefetch next V tile rows
            __builtin_prefetch(vT + (size_t)(hd + lane) * N + n0 + 32, 0, 1);

        // pull K B-fragments out of LDS (ds_load_b128 pairs)
        BF16x16 kb[2][2];
#pragma unroll
        for (int t = 0; t < 2; ++t)
#pragma unroll
            for (int c = 0; c < 2; ++c) {
                const __bf16* kr = kt + (t * 16 + lm) * 64 + c * 32 + g * 16;
                kb[t][c].h[0] = *(const v8bf*)(kr);
                kb[t][c].h[1] = *(const v8bf*)(kr + 8);
            }

        // ----- S = Q K^T, two 16-col sub-tiles -----------------------------
        F32x8 s[2];
#pragma unroll
        for (int t = 0; t < 2; ++t) {
#pragma unroll
            for (int r = 0; r < 8; ++r) s[t].e[r] = 0.f;
#pragma unroll
            for (int c = 0; c < 2; ++c)
                s[t].v = __builtin_amdgcn_wmma_f32_16x16x32_bf16(
                    false, qa[c].v, false, kb[t][c].v, (short)0, s[t].v,
                    false, false);
        }

        // ----- scale + label mask ------------------------------------------
        int nl0 = lbl1[n0 + lm];
        int nl1 = lbl1[n0 + 16 + lm];
#pragma unroll
        for (int r = 0; r < 8; ++r) {
            float s0 = s[0].e[r] * 0.25f;
            float s1 = s[1].e[r] * 0.25f;
            s[0].e[r] = (mylbl[r] == nl0) ? s0 : -1e30f;
            s[1].e[r] = (mylbl[r] == nl1) ? s1 : -1e30f;
        }

        // ----- online softmax ----------------------------------------------
#pragma unroll
        for (int r = 0; r < 8; ++r) {
            float mx = fmaxf(s[0].e[r], s[1].e[r]);
#pragma unroll
            for (int off = 8; off >= 1; off >>= 1)
                mx = fmaxf(mx, __shfl_xor(mx, off, 32));
            float newmax = fmaxf(rowmax[r], mx);
            float corr = __expf(rowmax[r] - newmax);
            rowmax[r] = newmax;
            rowsum[r] *= corr;
#pragma unroll
            for (int t = 0; t < 4; ++t) o[t].e[r] *= corr;
            float p0 = __expf(s[0].e[r] - newmax);
            float p1 = __expf(s[1].e[r] - newmax);
            s[0].e[r] = p0;
            s[1].e[r] = p1;
            float ps = p0 + p1;
#pragma unroll
            for (int off = 8; off >= 1; off >>= 1)
                ps += __shfl_xor(ps, off, 32);
            rowsum[r] += ps;
        }

        // ----- restripe P through LDS (C layout -> A layout) ---------------
#pragma unroll
        for (int t = 0; t < 2; ++t)
#pragma unroll
            for (int r = 0; r < 8; ++r)
                lds_p[(r + g * 8) * 32 + t * 16 + lm] = (__bf16)s[t].e[r];
        __syncthreads();

        BF16x16 pa;
        {
            const __bf16* pr = lds_p + lm * 32;
            pa.h[0] = *(const v8bf*)(pr + g * 16);
            pa.h[1] = *(const v8bf*)(pr + g * 16 + 8);
        }

        // ----- O += P * V; V fragments contiguous from transposed copy -----
#pragma unroll
        for (int t = 0; t < 4; ++t) {
            BF16x16 vb;   // B(K,N) = vT[hd+t*16+N][n0 + K], K = g*16+i
            const __bf16* vr = vT + (size_t)(hd + t * 16 + lm) * N + n0 + g * 16;
            vb.h[0] = *(const v8bf*)(vr);
            vb.h[1] = *(const v8bf*)(vr + 8);
            o[t].v = __builtin_amdgcn_wmma_f32_16x16x32_bf16(
                false, pa.v, false, vb.v, (short)0, o[t].v, false, false);
        }
        __syncthreads();
    }

    // ----- finalize --------------------------------------------------------
#pragma unroll
    for (int t = 0; t < 4; ++t)
#pragma unroll
        for (int r = 0; r < 8; ++r) {
            float val = o[t].e[r] / rowsum[r];
            ctx[(size_t)(m0 + r + g * 8) * D_MODEL + hd + t * 16 + lm] = (__bf16)val;
        }
}

// ---------------------------------------------------------------------------
// Kernel 3: out = LN(query + ctx @ Wf + bf) * gamma + beta.  grid = (M/16).
// WfT is the transposed bf16 copy, so B-fragments are contiguous.
// ---------------------------------------------------------------------------
__global__ __launch_bounds__(32)
void final_kernel(const __bf16* __restrict__ ctx,    // [M,256] bf16
                  const float*  __restrict__ query,  // [M,256] f32
                  const __bf16* __restrict__ WfT,    // [256,256] o-major bf16
                  const float*  __restrict__ bfv,    // [256]
                  const float*  __restrict__ gamma,  // [256]
                  const float*  __restrict__ beta,   // [256]
                  float*        __restrict__ out,    // [M,256] f32
                  int M)
{
    __shared__ __attribute__((aligned(16))) float lds_res[16 * D_MODEL]; // 16KB

    const int lane = threadIdx.x;
    const int g  = lane >> 4;
    const int lm = lane & 15;
    const int m0 = blockIdx.x * 16;

    for (int c0 = 0; c0 < D_MODEL; c0 += 16) {
        F32x8 acc;
#pragma unroll
        for (int r = 0; r < 8; ++r) acc.e[r] = 0.f;
        for (int k0 = 0; k0 < D_MODEL; k0 += 32) {
            BF16x16 a, bm;
            const __bf16* ar = ctx + (size_t)(m0 + lm) * D_MODEL + k0;
            a.h[0] = *(const v8bf*)(ar + g * 16);
            a.h[1] = *(const v8bf*)(ar + g * 16 + 8);
            const __bf16* wr = WfT + (size_t)(c0 + lm) * D_MODEL + k0 + g * 16;
            bm.h[0] = *(const v8bf*)(wr);
            bm.h[1] = *(const v8bf*)(wr + 8);
            acc.v = __builtin_amdgcn_wmma_f32_16x16x32_bf16(
                false, a.v, false, bm.v, (short)0, acc.v, false, false);
        }
#pragma unroll
        for (int r = 0; r < 8; ++r) {
            int m = r + g * 8;
            float val = acc.e[r] + bfv[c0 + lm] +
                        query[(size_t)(m0 + m) * D_MODEL + c0 + lm];
            lds_res[m * D_MODEL + c0 + lm] = val;
        }
    }
    __syncthreads();

    for (int m = 0; m < 16; ++m) {
        float s = 0.f, s2 = 0.f;
#pragma unroll
        for (int j = 0; j < 8; ++j) {
            float x = lds_res[m * D_MODEL + lane + j * 32];
            s += x; s2 += x * x;
        }
#pragma unroll
        for (int off = 16; off >= 1; off >>= 1) {
            s  += __shfl_xor(s,  off, 32);
            s2 += __shfl_xor(s2, off, 32);
        }
        float mu  = s / (float)D_MODEL;
        float var = s2 / (float)D_MODEL - mu * mu;
        float inv = rsqrtf(var + 1e-5f);
#pragma unroll
        for (int j = 0; j < 8; ++j) {
            int c = lane + j * 32;
            float x = lds_res[m * D_MODEL + c];
            out[(size_t)(m0 + m) * D_MODEL + c] = (x - mu) * inv * gamma[c] + beta[c];
        }
    }
}

// ---------------------------------------------------------------------------
extern "C" void kernel_launch(void* const* d_in, const int* in_sizes, int n_in,
                              void* d_out, int out_size, void* d_ws, size_t ws_size,
                              hipStream_t stream) {
    const float* key_  = (const float*)d_in[0];
    const float* value = (const float*)d_in[1];
    const float* query = (const float*)d_in[2];
    const int*   lbl1  = (const int*)d_in[3];
    const int*   lbl2  = (const int*)d_in[4];
    const float* Wq    = (const float*)d_in[5];
    const float* bq    = (const float*)d_in[6];
    const float* Wk    = (const float*)d_in[7];
    const float* bk    = (const float*)d_in[8];
    const float* Wv    = (const float*)d_in[9];
    const float* bv    = (const float*)d_in[10];
    const float* Wf    = (const float*)d_in[11];
    const float* bfv   = (const float*)d_in[12];
    const float* gamma = (const float*)d_in[13];
    const float* beta  = (const float*)d_in[14];
    float* out = (float*)d_out;

    const int N = in_sizes[0] / D_MODEL;
    const int M = in_sizes[2] / D_MODEL;
    const size_t WSLICE = (size_t)D_MODEL * D_MODEL;   // 65536 elems

    char* ws = (char*)d_ws;
    size_t off = 0;
    __bf16* qp  = (__bf16*)(ws + off); off += (size_t)M * D_MODEL * 2;
    __bf16* kp  = (__bf16*)(ws + off); off += (size_t)N * D_MODEL * 2;
    __bf16* vTp = (__bf16*)(ws + off); off += (size_t)N * D_MODEL * 2;
    __bf16* cx  = (__bf16*)(ws + off); off += (size_t)M * D_MODEL * 2;
    __bf16* xq  = (__bf16*)(ws + off); off += (size_t)M * D_MODEL * 2;
    __bf16* xk  = (__bf16*)(ws + off); off += (size_t)N * D_MODEL * 2;
    __bf16* xv  = (__bf16*)(ws + off); off += (size_t)N * D_MODEL * 2;
    __bf16* WTq = (__bf16*)(ws + off); off += NLBL * WSLICE * 2;
    __bf16* WTk = (__bf16*)(ws + off); off += NLBL * WSLICE * 2;
    __bf16* WTv = (__bf16*)(ws + off); off += NLBL * WSLICE * 2;
    __bf16* WTf = (__bf16*)(ws + off); off += WSLICE * 2;

    dim3 blk32(32, 1, 1);
    dim3 blk256(256, 1, 1);

    // one-time converts (activations + transposed bf16 weights)
    cvt_bf16_kernel<<<dim3((M * D_MODEL / 8 + 255) / 256), blk256, 0, stream>>>(
        query, xq, M * D_MODEL);
    cvt_bf16_kernel<<<dim3((N * D_MODEL / 8 + 255) / 256), blk256, 0, stream>>>(
        key_, xk, N * D_MODEL);
    cvt_bf16_kernel<<<dim3((N * D_MODEL / 8 + 255) / 256), blk256, 0, stream>>>(
        value, xv, N * D_MODEL);
    transpose_cvt_kernel<<<dim3(16, 16, NLBL), blk32, 0, stream>>>(Wq, WTq);
    transpose_cvt_kernel<<<dim3(16, 16, NLBL), blk32, 0, stream>>>(Wk, WTk);
    transpose_cvt_kernel<<<dim3(16, 16, NLBL), blk32, 0, stream>>>(Wv, WTv);
    transpose_cvt_kernel<<<dim3(16, 16, 1),    blk32, 0, stream>>>(Wf, WTf);

    // per-label projections (V stored transposed for attention)
    proj_label_kernel<<<dim3(M / 16, D_MODEL / 16, NLBL), blk32, 0, stream>>>(
        xq, lbl2, WTq, bq, qp, M, 0);
    proj_label_kernel<<<dim3(N / 16, D_MODEL / 16, NLBL), blk32, 0, stream>>>(
        xk, lbl1, WTk, bk, kp, N, 0);
    proj_label_kernel<<<dim3(N / 16, D_MODEL / 16, NLBL), blk32, 0, stream>>>(
        xv, lbl1, WTv, bv, vTp, N, 1);

    attn_kernel<<<dim3(M / 16, NHEAD), blk32, 0, stream>>>(
        qp, kp, vTp, lbl2, lbl1, cx, M, N);
    final_kernel<<<dim3(M / 16), blk32, 0, stream>>>(
        cx, query, WTf, bfv, gamma, beta, out, M);
}
